// Net_91225105367817
// MI455X (gfx1250) — matrix operand order repaired and checked by reading, not compiled, not run on hardware
//
#include <hip/hip_runtime.h>
#include <hip/hip_bf16.h>

// ---------------------------------------------------------------------------
// ParticleNet-style GNN on gfx1250 (wave32, WMMA 16x16x32 f16)
//   encoders -> 3x fused EdgeConv (WMMA Gram + top8 + WMMA edge MLP + max)
//   -> mean pool -> output MLP.
// All intermediates kept in f16 in d_ws.
// ---------------------------------------------------------------------------

typedef __attribute__((ext_vector_type(16))) _Float16 v16h;
typedef __attribute__((ext_vector_type(8)))  _Float16 v8h;
typedef __attribute__((ext_vector_type(8)))  float    v8f;

#define Bn    256
#define NSV   64
#define NTRK  512
#define NPFC  512
#define KNN   8
#define Hd    32

__device__ __forceinline__ float elu_f(float x) {
    return x > 0.f ? x : (__expf(x) - 1.f);
}

// 16 contiguous halves (32B, 16B-aligned) -> v16h
__device__ __forceinline__ v16h ld16(const _Float16* p) {
    v8h lo = *(const v8h*)p;
    v8h hi = *(const v8h*)(p + 8);
    v16h r;
#pragma unroll
    for (int i = 0; i < 8; ++i) { r[i] = lo[i]; r[8 + i] = hi[i]; }
    return r;
}

// A fragment, 16x32 f16, row-major with `stride` halves per row.
// Lane L: m = L%16; holds K = k0..k0+15 contiguous, k0 = (L<16 ? 0 : 16).
__device__ __forceinline__ v16h a_frag(const _Float16* base, int stride, int lane) {
    int m  = lane & 15;
    int k0 = (lane < 16) ? 0 : 16;
    const _Float16* row = base + m * stride;
    v8h lo = *(const v8h*)(row + k0);
    v8h hi = *(const v8h*)(row + k0 + 8);
    v16h r;
#pragma unroll
    for (int i = 0; i < 8; ++i) { r[i] = lo[i]; r[8 + i] = hi[i]; }
    return r;
}

// B fragment from K-major storage bt[n][k] (stride halves per n-row).
// Lane L: n = L%16; holds K = (L<16 ? 0..15 : 16..31), contiguous.
__device__ __forceinline__ v16h b_frag(const _Float16* bt, int stride, int lane) {
    int n  = lane & 15;
    int k0 = (lane < 16) ? 0 : 16;
    return ld16(bt + n * stride + k0);
}

__device__ __forceinline__ v8f wmma_f16(v16h a, v16h b, v8f c) {
    return __builtin_amdgcn_wmma_f32_16x16x32_f16(
        /*neg_a=*/false, a, /*neg_b=*/false, b,
        /*c_mod=*/(short)0, c, /*reuse_a=*/false, /*reuse_b=*/false);
}

// ---------------------------------------------------------------------------
// Encoder: 2-layer MLP (fan->32->32, ELU), 16 nodes per single-wave block.
// ---------------------------------------------------------------------------
__global__ __launch_bounds__(32) void encode_kernel(
    const float* __restrict__ x, int fan,
    const float* __restrict__ W1, const float* __restrict__ b1,
    const float* __restrict__ W2, const float* __restrict__ b2,
    _Float16* __restrict__ out)
{
    __shared__ __align__(16) _Float16 xa[16 * 32];   // node features, K-padded
    __shared__ __align__(16) _Float16 w1t[32 * 32];  // w1t[n][k] = W1[k][n]
    __shared__ __align__(16) _Float16 w2t[32 * 32];
    __shared__ __align__(16) _Float16 hb[16 * 32];   // hidden activations

    const int lane  = threadIdx.x;
    const long node0 = (long)blockIdx.x * 16;

    for (int idx = lane; idx < 16 * 32; idx += 32) {
        int r = idx >> 5, k = idx & 31;
        xa[idx] = (k < fan) ? (_Float16)x[(node0 + r) * fan + k] : (_Float16)0.f;
    }
    for (int idx = lane; idx < 32 * 32; idx += 32) {
        int n = idx >> 5, k = idx & 31;
        w1t[idx] = (k < fan) ? (_Float16)W1[k * 32 + n] : (_Float16)0.f;
        w2t[idx] = (_Float16)W2[k * 32 + n];
    }
    __syncthreads();

    const int n  = lane & 15;
    const int mb = (lane < 16) ? 0 : 8;

    // layer 1: [16 x fan(pad32)] @ [fan x 32]
    v16h a = a_frag(xa, 32, lane);
    v8f c0 = {}, c1 = {};
    c0 = wmma_f16(a, b_frag(w1t,           32, lane), c0);
    c1 = wmma_f16(a, b_frag(w1t + 16 * 32, 32, lane), c1);
    float bb0 = b1[n], bb1 = b1[n + 16];
#pragma unroll
    for (int r = 0; r < 8; ++r) {
        hb[(mb + r) * 32 + n]      = (_Float16)elu_f(c0[r] + bb0);
        hb[(mb + r) * 32 + n + 16] = (_Float16)elu_f(c1[r] + bb1);
    }
    __syncthreads();

    // layer 2: [16 x 32] @ [32 x 32]
    a = a_frag(hb, 32, lane);
    v8f d0 = {}, d1 = {};
    d0 = wmma_f16(a, b_frag(w2t,           32, lane), d0);
    d1 = wmma_f16(a, b_frag(w2t + 16 * 32, 32, lane), d1);
    bb0 = b2[n]; bb1 = b2[n + 16];
#pragma unroll
    for (int r = 0; r < 8; ++r) {
        out[(node0 + mb + r) * 32 + n]      = (_Float16)elu_f(d0[r] + bb0);
        out[(node0 + mb + r) * 32 + n + 16] = (_Float16)elu_f(d1[r] + bb1);
    }
}

// ---------------------------------------------------------------------------
// Fused EdgeConv: per (event, 16-dst-node tile) single-wave workgroup.
//  1) WMMA Gram matrix G = Xdst * Xsrc^T (tiles of 16 src), kNN score in LDS
//  2) serial top-8 select per dst row (lanes 0..15)
//  3) gather + build edge features [x_i, x_j - x_i] (128 edges x 64) in LDS
//     (vectorized v8h: b128 loads, packed f16 subtract, b128 LDS stores)
//  4) WMMA edge MLP (K=64 as 2x K32, N=32 as 2x N16), ELU, max over 8 nbrs
// ---------------------------------------------------------------------------
__global__ __launch_bounds__(32) void edge_conv_kernel(
    const _Float16* __restrict__ src, int Ns,
    const _Float16* __restrict__ dst,
    const float* __restrict__ W,      // [64][32] row-major (k-major)
    const float* __restrict__ bias,   // [32]
    _Float16* __restrict__ out)
{
    __shared__ __align__(16) float    score[16 * 512];  // sized for max Ns
    __shared__ __align__(16) _Float16 wt[32 * 64];      // wt[n][k] = W[k][n]
    __shared__ __align__(16) _Float16 ebuf[128 * 64];   // edge features
    __shared__ int topk[16 * KNN];

    const int lane = threadIdx.x;
    const int b    = blockIdx.y;
    const int tile = blockIdx.x;
    const int Nd   = NTRK;

    const _Float16* dst_base = dst + ((long)b * Nd + tile * 16) * Hd;
    const _Float16* src_base = src + (long)b * Ns * Hd;

    for (int idx = lane; idx < 32 * 64; idx += 32) {
        int n = idx / 64, k = idx % 64;
        wt[idx] = (_Float16)W[k * 32 + n];
    }

    const int n  = lane & 15;
    const int mb = (lane < 16) ? 0 : 8;
    const int k0 = (lane < 16) ? 0 : 16;

    // --- 1) distances via WMMA ---
    v16h adst = a_frag(dst_base, 32, lane);
    for (int s = 0; s < Ns; s += 16) {
        // prefetch next src tile into cache while we do WMMA + score writeback
        if (s + 16 < Ns)
            __builtin_prefetch(src_base + (s + 16 + n) * Hd + k0, 0, 1);
        const _Float16* srow = src_base + (s + n) * Hd;
        v16h bfr = ld16(srow + k0);            // B fragment: src^T tile
        float pn = 0.f;
#pragma unroll
        for (int i = 0; i < 16; ++i) { float v = (float)bfr[i]; pn += v * v; }
        float srcnorm = pn + __shfl_xor(pn, 16, 32);
        v8f c = {};
        c = wmma_f16(adst, bfr, c);
#pragma unroll
        for (int r = 0; r < 8; ++r)
            score[(mb + r) * Ns + s + n] = srcnorm - 2.f * c[r];
    }
    __syncthreads();

    // --- 2) top-8 smallest per dst row ---
    if (lane < 16) {
        const float* row = score + lane * Ns;
        float bs[KNN]; int bi[KNN];
#pragma unroll
        for (int i = 0; i < KNN; ++i) { bs[i] = row[i]; bi[i] = i; }
        int w = 0;
#pragma unroll
        for (int i = 1; i < KNN; ++i) if (bs[i] > bs[w]) w = i;
        for (int j = KNN; j < Ns; ++j) {
            float v = row[j];
            if (v < bs[w]) {
                bs[w] = v; bi[w] = j;
                w = 0;
#pragma unroll
                for (int i = 1; i < KNN; ++i) if (bs[i] > bs[w]) w = i;
            }
        }
#pragma unroll
        for (int i = 0; i < KNN; ++i) topk[lane * KNN + i] = bi[i];
    }
    __syncthreads();

    // --- 3) gather + edge features, vectorized v8h (4 chunks of 8 halves) ---
    for (int e = lane; e < 128; e += 32) {
        int dl = e >> 3;
        int j  = topk[e];                      // topk[dl*8 + (e&7)]
        const v8h* xi = (const v8h*)(dst_base + dl * Hd);
        const v8h* xj = (const v8h*)(src_base + j * Hd);
        v8h* eb = (v8h*)(ebuf + e * 64);
#pragma unroll
        for (int c = 0; c < 4; ++c) {
            v8h a  = xi[c];
            v8h dj = xj[c] - a;                // packed f16 subtract
            eb[c]     = a;
            eb[4 + c] = dj;
        }
    }
    __syncthreads();

    // --- 4) edge MLP + ELU + max over neighbors ---
    const float bb0 = bias[n], bb1 = bias[n + 16];
    for (int mt = 0; mt < 8; ++mt) {
        v8f c0 = {}, c1 = {};
#pragma unroll
        for (int kc = 0; kc < 64; kc += 32) {
            v16h a = a_frag(ebuf + mt * 16 * 64 + kc, 64, lane);
            c0 = wmma_f16(a, b_frag(wt + kc,           64, lane), c0);
            c1 = wmma_f16(a, b_frag(wt + 16 * 64 + kc, 64, lane), c1);
        }
        // rows 0..7 (lanes<16) = 8 edges of dst 2*mt; rows 8..15 = dst 2*mt+1
        float m0 = -3.4e38f, m1 = -3.4e38f;
#pragma unroll
        for (int r = 0; r < 8; ++r) {
            m0 = fmaxf(m0, elu_f(c0[r] + bb0));
            m1 = fmaxf(m1, elu_f(c1[r] + bb1));
        }
        int dl = 2 * mt + ((lane < 16) ? 0 : 1);
        long o = ((long)b * Nd + tile * 16 + dl) * Hd + n;
        out[o]      = (_Float16)m0;
        out[o + 16] = (_Float16)m1;
    }
}

// ---------------------------------------------------------------------------
// Mean pool over 512 trk nodes + output MLP (32->32 ELU ->1 sigmoid).
// ---------------------------------------------------------------------------
__global__ __launch_bounds__(32) void pool_out_kernel(
    const _Float16* __restrict__ f3,
    const float* __restrict__ W1, const float* __restrict__ b1,
    const float* __restrict__ W2, const float* __restrict__ b2,
    float* __restrict__ out, long long* __restrict__ idx_out)
{
    __shared__ float pool[32];
    __shared__ float y1[32];
    const int lane = threadIdx.x;   // = channel
    const int b    = blockIdx.x;
    const _Float16* base = f3 + (long)b * NTRK * Hd;

    float acc = 0.f;
    for (int v = 0; v < NTRK; ++v) acc += (float)base[v * Hd + lane];
    pool[lane] = acc * (1.f / (float)NTRK);
    __syncthreads();

    float a1 = b1[lane];
    for (int h = 0; h < 32; ++h) a1 += pool[h] * W1[h * 32 + lane];
    y1[lane] = elu_f(a1);
    __syncthreads();

    float part = y1[lane] * W2[lane];
#pragma unroll
    for (int off = 16; off; off >>= 1) part += __shfl_xor(part, off, 32);
    if (lane == 0) {
        float z = part + b2[0];
        out[b] = 1.f / (1.f + __expf(-z));
        idx_out[b] = (long long)b;
    }
}

// ---------------------------------------------------------------------------
extern "C" void kernel_launch(void* const* d_in, const int* in_sizes, int n_in,
                              void* d_out, int out_size, void* d_ws, size_t ws_size,
                              hipStream_t stream) {
    (void)in_sizes; (void)n_in; (void)out_size; (void)ws_size;

    const float* x_sv   = (const float*)d_in[0];
    const float* x_trk  = (const float*)d_in[1];
    const float* x_pfc  = (const float*)d_in[2];
    // d_in[3..5]: batch index arrays (implicit in dense layout, unused)
    const float* sv_W1  = (const float*)d_in[6];
    const float* sv_b1  = (const float*)d_in[7];
    const float* sv_W2  = (const float*)d_in[8];
    const float* sv_b2  = (const float*)d_in[9];
    const float* trk_W1 = (const float*)d_in[10];
    const float* trk_b1 = (const float*)d_in[11];
    const float* trk_W2 = (const float*)d_in[12];
    const float* trk_b2 = (const float*)d_in[13];
    const float* pfc_W1 = (const float*)d_in[14];
    const float* pfc_b1 = (const float*)d_in[15];
    const float* pfc_W2 = (const float*)d_in[16];
    const float* pfc_b2 = (const float*)d_in[17];
    const float* conv_W = (const float*)d_in[18];
    const float* conv_b = (const float*)d_in[19];
    const float* out_W1 = (const float*)d_in[20];
    const float* out_b1 = (const float*)d_in[21];
    const float* out_W2 = (const float*)d_in[22];
    const float* out_b2 = (const float*)d_in[23];

    // f16 intermediates in workspace (~41 MB)
    _Float16* sv_h  = (_Float16*)d_ws;
    _Float16* trk_h = sv_h  + (size_t)Bn * NSV  * Hd;
    _Float16* pfc_h = trk_h + (size_t)Bn * NTRK * Hd;
    _Float16* f1_h  = pfc_h + (size_t)Bn * NPFC * Hd;
    _Float16* f2_h  = f1_h  + (size_t)Bn * NTRK * Hd;
    _Float16* f3_h  = f2_h  + (size_t)Bn * NTRK * Hd;

    encode_kernel<<<dim3((Bn * NSV ) / 16), dim3(32), 0, stream>>>(
        x_sv, 14, sv_W1, sv_b1, sv_W2, sv_b2, sv_h);
    encode_kernel<<<dim3((Bn * NTRK) / 16), dim3(32), 0, stream>>>(
        x_trk, 30, trk_W1, trk_b1, trk_W2, trk_b2, trk_h);
    encode_kernel<<<dim3((Bn * NPFC) / 16), dim3(32), 0, stream>>>(
        x_pfc, 10, pfc_W1, pfc_b1, pfc_W2, pfc_b2, pfc_h);

    edge_conv_kernel<<<dim3(NTRK / 16, Bn), dim3(32), 0, stream>>>(
        sv_h, NSV, trk_h, conv_W, conv_b, f1_h);
    edge_conv_kernel<<<dim3(NTRK / 16, Bn), dim3(32), 0, stream>>>(
        pfc_h, NPFC, trk_h, conv_W, conv_b, f2_h);
    edge_conv_kernel<<<dim3(NTRK / 16, Bn), dim3(32), 0, stream>>>(
        f1_h, NTRK, f2_h, conv_W, conv_b, f3_h);

    float*     out_f = (float*)d_out;
    long long* out_i = (long long*)(out_f + Bn);   // int64 arange after B floats
    pool_out_kernel<<<dim3(Bn), dim3(32), 0, stream>>>(
        f3_h, out_W1, out_b1, out_W2, out_b2, out_f, out_i);
}
